// Decoder_78022375899610
// MI455X (gfx1250) — matrix operand order repaired and checked by reading, not compiled, not run on hardware
//
#include <hip/hip_runtime.h>
#include <hip/hip_bf16.h>

// ---------------------------------------------------------------------------
// CDNA5 (gfx1250) transformer decoder. All GEMMs + attention use
// v_wmma_f32_16x16x32_f16 (wave32). ~310 GFLOP vs ~100MB weights (resident in
// 192MB L2) => compute bound: f16 WMMA inputs, f32 accumulation.
// Row-major global->LDS tile copies use GLOBAL_LOAD_ASYNC_TO_LDS_B128
// (ASYNCcnt + s_wait_asynccnt), GEMM k-loop is LDS double-buffered.
// ---------------------------------------------------------------------------

typedef __attribute__((ext_vector_type(16))) _Float16     v16h;
typedef __attribute__((ext_vector_type(8)))  float        v8f;
typedef __attribute__((ext_vector_type(8)))  unsigned int v8u;

#define WMMA16(A, B, C) \
  __builtin_amdgcn_wmma_f32_16x16x32_f16(false, (A), false, (B), (short)0, (C), false, false)

// --- CDNA5 async global->LDS copy (16B per lane), tracked by ASYNCcnt ------
__device__ __forceinline__ void async_g2l_b128(unsigned lds_addr, const void* g) {
  asm volatile("global_load_async_to_lds_b128 %0, %1, off"
               :: "v"(lds_addr), "v"((unsigned long long)g)
               : "memory");
}
__device__ __forceinline__ void wait_asynccnt0() {
  asm volatile("s_wait_asynccnt 0x0" ::: "memory");
}
__device__ __forceinline__ unsigned lds_addr_of(const void* p) {
  return (unsigned)(unsigned long long)p;  // low 32 bits of LDS aperture addr
}

// 16-bit A-matrix 16x32 fragment layout (ISA 7.12.2):
//   lane m = lane&15, half = lane>>4; VGPR v holds K pair:
//   k0 = (v>>2)*16 + half*8 + (v&3)*2
__device__ __forceinline__ int a_frag_k(int v, int half) {
  return ((v >> 2) << 4) + (half << 3) + ((v & 3) << 1);
}

__device__ __forceinline__ v16h load_a_frag(const _Float16* row, int half) {
  v8u u;
#pragma unroll
  for (int v = 0; v < 8; ++v)
    u[v] = *(const unsigned int*)(row + a_frag_k(v, half));
  return __builtin_bit_cast(v16h, u);
}

// 16-bit B-matrix 32x16 fragment: lane n = lane&15, half = lane>>4;
// VGPR v holds K pair k0 = a_frag_k(v, half). `row` = n-th row of a [N][K] tile.
__device__ __forceinline__ v16h load_b_frag(const _Float16* row, int half) {
  v8u u;
#pragma unroll
  for (int v = 0; v < 8; ++v)
    u[v] = *(const unsigned int*)(row + a_frag_k(v, half));
  return __builtin_bit_cast(v16h, u);
}

// ---------------------------------------------------------------------------
// f32 -> f16 convert (weights / memory prologue)
// ---------------------------------------------------------------------------
__global__ void cvt_f32_f16_k(const float* __restrict__ in,
                              _Float16* __restrict__ out, size_t n) {
  size_t i = (size_t)blockIdx.x * blockDim.x + threadIdx.x;
  size_t stride = (size_t)gridDim.x * blockDim.x;
  for (; i < n; i += stride) out[i] = (_Float16)in[i];
}

// ---------------------------------------------------------------------------
// LayerNorm over D=512: one wave per row, wave32 shuffle reductions.
// ---------------------------------------------------------------------------
template <bool OUT16>
__global__ void __launch_bounds__(256)
layernorm_k(const float* __restrict__ x, const float* __restrict__ g,
            const float* __restrict__ b, _Float16* __restrict__ o16,
            float* __restrict__ o32) {
  const int row  = blockIdx.x * 8 + (threadIdx.x >> 5);
  const int lane = threadIdx.x & 31;
  const float* xr = x + (size_t)row * 512;
  float vals[16];
  float s = 0.f;
#pragma unroll
  for (int i = 0; i < 16; ++i) { vals[i] = xr[lane + i * 32]; s += vals[i]; }
#pragma unroll
  for (int m = 16; m >= 1; m >>= 1) s += __shfl_xor(s, m, 32);
  const float mean = s * (1.f / 512.f);
  float vs = 0.f;
#pragma unroll
  for (int i = 0; i < 16; ++i) { float d = vals[i] - mean; vs += d * d; }
#pragma unroll
  for (int m = 16; m >= 1; m >>= 1) vs += __shfl_xor(vs, m, 32);
  const float rstd = rsqrtf(vs * (1.f / 512.f) + 1e-5f);
#pragma unroll
  for (int i = 0; i < 16; ++i) {
    int c = lane + i * 32;
    float y = (vals[i] - mean) * rstd * g[c] + b[c];
    if (OUT16) o16[(size_t)row * 512 + c] = (_Float16)y;
    else       o32[(size_t)row * 512 + c] = y;
  }
}

// ---------------------------------------------------------------------------
// WMMA GEMM: C[M,N] = A[M,K] * B[K,N] + bias (+resid) (+relu)
// 128x128 block tile, BK=32, compile-time N/K (strides fold to immediates),
// LDS double-buffered; A tile copied with async global->LDS b128; B tile
// loaded to VGPRs, pair-packed, stored transposed [n][k] as ds_store_b32.
// 8 waves in a 2(M) x 4(N) grid; each wave 4x2 16x16 WMMA tiles.
// ---------------------------------------------------------------------------
template <int N, int K, bool RELU, bool RESID, bool OUT16>
__global__ void __launch_bounds__(256)
gemm_wmma_k(const _Float16* __restrict__ A, const _Float16* __restrict__ Bw,
            const float* __restrict__ bias, const float* __restrict__ resid,
            _Float16* __restrict__ o16, float* __restrict__ o32) {
  constexpr int LDA = 40, LDB = 40;  // +8 f16 pad vs 64-bank LDS
  __shared__ _Float16 As[2][128 * LDA];
  __shared__ _Float16 Bs[2][128 * LDB];  // transposed: [n][k]

  const int tid  = threadIdx.x;
  const int lane = tid & 31, wave = tid >> 5;
  const int half = lane >> 4, nl = lane & 15;
  const int wm = wave & 1, wn = wave >> 1;
  const int mtile = blockIdx.x * 128, ntile = blockIdx.y * 128;

  v8f acc[4][2];
#pragma unroll
  for (int i = 0; i < 4; ++i)
#pragma unroll
    for (int j = 0; j < 2; ++j)
      acc[i][j] = (v8f){0.f, 0.f, 0.f, 0.f, 0.f, 0.f, 0.f, 0.f};

  // --- tile movers -----------------------------------------------------
  const int ar  = tid >> 1, aseg = (tid & 1) << 4;      // A: 16 f16 per thread
  const int bj  = tid >> 4, bng  = tid & 15;            // B: rows 2bj,2bj+1 x 8 cols
  uint4 br0, br1;

  auto issue_A = [&](int buf, int k0) {
    const _Float16* g = A + (size_t)(mtile + ar) * K + k0 + aseg;
    unsigned l = lds_addr_of(&As[buf][ar * LDA + aseg]);
    async_g2l_b128(l, g);
    async_g2l_b128(l + 16, g + 8);
  };
  auto issue_B = [&](int k0) {
    br0 = *(const uint4*)(Bw + (size_t)(k0 + 2 * bj) * N + ntile + bng * 8);
    br1 = *(const uint4*)(Bw + (size_t)(k0 + 2 * bj + 1) * N + ntile + bng * 8);
  };
  auto store_B = [&](int buf) {
    _Float16* dst = &Bs[buf][0];
#pragma unroll
    for (int w = 0; w < 4; ++w) {
      unsigned a = (&br0.x)[w], b = (&br1.x)[w];
      unsigned lo = (a & 0xffffu) | (b << 16);
      unsigned hi = (a >> 16) | (b & 0xffff0000u);
      *(unsigned*)(dst + (bng * 8 + 2 * w) * LDB + 2 * bj)     = lo;
      *(unsigned*)(dst + (bng * 8 + 2 * w + 1) * LDB + 2 * bj) = hi;
    }
  };

  constexpr int NSTEP = K / 32;
  issue_A(0, 0);
  issue_B(0);
  store_B(0);
  wait_asynccnt0();
  __syncthreads();

  for (int step = 0; step < NSTEP; ++step) {
    const int buf = step & 1;
    if (step + 1 < NSTEP) {           // prefetch next tile while computing
      issue_A(buf ^ 1, (step + 1) * 32);
      issue_B((step + 1) * 32);
    }
    const _Float16* Ab = As[buf];
    const _Float16* Bb = Bs[buf];
    v16h af[4], bf[2];
#pragma unroll
    for (int mi = 0; mi < 4; ++mi)
      af[mi] = load_a_frag(Ab + (wm * 64 + mi * 16 + nl) * LDA, half);
#pragma unroll
    for (int ni = 0; ni < 2; ++ni)
      bf[ni] = load_b_frag(Bb + (wn * 32 + ni * 16 + nl) * LDB, half);
#pragma unroll
    for (int mi = 0; mi < 4; ++mi)
#pragma unroll
      for (int ni = 0; ni < 2; ++ni)
        acc[mi][ni] = WMMA16(af[mi], bf[ni], acc[mi][ni]);
    if (step + 1 < NSTEP) {
      store_B(buf ^ 1);
      wait_asynccnt0();
    }
    __syncthreads();
  }

  // C/D layout: lane -> n = lane&15; VGPR v -> m = half*8 + v
#pragma unroll
  for (int mi = 0; mi < 4; ++mi)
#pragma unroll
    for (int ni = 0; ni < 2; ++ni) {
      int n = ntile + wn * 32 + ni * 16 + nl;
      float bv = bias[n];
#pragma unroll
      for (int v = 0; v < 8; ++v) {
        int m = mtile + wm * 64 + mi * 16 + half * 8 + v;
        float val = acc[mi][ni][v] + bv;
        if (RELU)  val = fmaxf(val, 0.f);
        if (RESID) val += resid[(size_t)m * N + n];
        if (OUT16) o16[(size_t)m * N + n] = (_Float16)val;
        else       o32[(size_t)m * N + n] = val;
      }
    }
}

// ---------------------------------------------------------------------------
// Flash attention (WMMA): block = 4 waves x 16 Q rows = 64 Q rows of one
// (b, h). DK = 64. 32-key K/V LDS tiles: K via async b128 copies, V loaded
// to VGPRs + pair-packed transposed stores. P repacked C->A layout via LDS
// with an explicit s_wait_dscnt. Causal mask analytic (target_mask is tril;
// source_mask all-ones).
// ---------------------------------------------------------------------------
template <bool CAUSAL, int SKV>
__global__ void __launch_bounds__(128)
flash_attn_k(const _Float16* __restrict__ Q, const _Float16* __restrict__ Kt,
             const _Float16* __restrict__ Vt, _Float16* __restrict__ O) {
  __shared__ _Float16 Ks[32 * 72];       // [key 32][dk 64] (+8 pad)
  __shared__ _Float16 Vs[64 * 40];       // transposed: [dv 64][key 32] (+8 pad)
  __shared__ _Float16 Ps[4 * 16 * 40];   // per-wave P: [m 16][k 32] (+8 pad)

  const int tid = threadIdx.x, lane = tid & 31, wave = tid >> 5;
  const int half = lane >> 4, nl = lane & 15;
  const int bidx = blockIdx.y >> 3, h = blockIdx.y & 7;
  const int q0 = blockIdx.x * 64 + wave * 16;
  const size_t qrow0 = (size_t)bidx * 512 + q0;   // rows in [B*T, 512]
  const size_t krow0 = (size_t)bidx * SKV;        // rows in [B*SKV, 512]
  const int col0 = h * 64;

  // Q fragments (16x64 = two 16x32 A-frags), straight from global.
  v16h qf[2];
  {
    const _Float16* qr = Q + (qrow0 + nl) * 512 + col0;
#pragma unroll
    for (int c = 0; c < 2; ++c) qf[c] = load_a_frag(qr + c * 32, half);
  }

  v8f o[4];
#pragma unroll
  for (int nt = 0; nt < 4; ++nt)
    o[nt] = (v8f){0.f, 0.f, 0.f, 0.f, 0.f, 0.f, 0.f, 0.f};
  float mrow[8], lrow[8];
#pragma unroll
  for (int v = 0; v < 8; ++v) { mrow[v] = -1e30f; lrow[v] = 0.f; }

  const int kr_ = tid >> 2, kseg = (tid & 3) << 4;  // K copy: 16 f16 per thread
  const int vj = tid >> 3, vng = tid & 7;           // V: rows 2vj,2vj+1 x 8 cols

  const int kend = CAUSAL ? (blockIdx.x * 64 + 64) : SKV;
  for (int kb = 0; kb < kend; kb += 32) {
    __syncthreads();   // previous tile fully consumed
    {  // K tile: async global->LDS (row-major copy)
      const _Float16* g = Kt + (krow0 + kb + kr_) * 512 + col0 + kseg;
      unsigned l = lds_addr_of(&Ks[kr_ * 72 + kseg]);
      async_g2l_b128(l, g);
      async_g2l_b128(l + 16, g + 8);
    }
    {  // V tile: pack key-pairs, store transposed as b32
      uint4 a = *(const uint4*)(Vt + (krow0 + kb + 2 * vj) * 512 + col0 + vng * 8);
      uint4 b = *(const uint4*)(Vt + (krow0 + kb + 2 * vj + 1) * 512 + col0 + vng * 8);
#pragma unroll
      for (int w = 0; w < 4; ++w) {
        unsigned x = (&a.x)[w], y = (&b.x)[w];
        unsigned lo = (x & 0xffffu) | (y << 16);
        unsigned hi = (x >> 16) | (y & 0xffff0000u);
        *(unsigned*)(Vs + (vng * 8 + 2 * w) * 40 + 2 * vj)     = lo;
        *(unsigned*)(Vs + (vng * 8 + 2 * w + 1) * 40 + 2 * vj) = hi;
      }
    }
    wait_asynccnt0();
    __syncthreads();

    // scores: S(16x32) = Q(16x64) x K^T(64x32) as 2 subtiles of 16 keys
    v8f sc[2];
#pragma unroll
    for (int sub = 0; sub < 2; ++sub) {
      v8f s8 = (v8f){0.f, 0.f, 0.f, 0.f, 0.f, 0.f, 0.f, 0.f};
      const _Float16* kr = Ks + (sub * 16 + nl) * 72;  // B[kk,s]=K[s,kk]: pairs contiguous
      v16h b0 = load_b_frag(kr, half);
      v16h b1 = load_b_frag(kr + 32, half);
      s8 = WMMA16(qf[0], b0, s8);
      s8 = WMMA16(qf[1], b1, s8);
      sc[sub] = s8;
    }

    // online softmax; row m = half*8 + v, key col = nl (+16 for subtile 1)
    _Float16* pw = Ps + wave * (16 * 40);
#pragma unroll
    for (int v = 0; v < 8; ++v) {
      float s0 = sc[0][v] * 0.125f;  // 1/sqrt(DK)
      float s1 = sc[1][v] * 0.125f;
      if (CAUSAL) {
        int q = q0 + half * 8 + v;
        if (kb + nl > q)      s0 = -1e30f;
        if (kb + 16 + nl > q) s1 = -1e30f;
      }
      float mv = fmaxf(s0, s1);
#pragma unroll
      for (int msk = 8; msk >= 1; msk >>= 1) mv = fmaxf(mv, __shfl_xor(mv, msk, 32));
      float mn = fmaxf(mrow[v], mv);
      float alpha = __expf(mrow[v] - mn);
      float p0 = __expf(s0 - mn), p1 = __expf(s1 - mn);
      float rs = p0 + p1;
#pragma unroll
      for (int msk = 8; msk >= 1; msk >>= 1) rs += __shfl_xor(rs, msk, 32);
      lrow[v] = lrow[v] * alpha + rs;
      mrow[v] = mn;
#pragma unroll
      for (int nt = 0; nt < 4; ++nt) o[nt][v] *= alpha;
      _Float16* pr = pw + (half * 8 + v) * 40;
      pr[nl]      = (_Float16)p0;
      pr[16 + nl] = (_Float16)p1;
    }
    // wave-local LDS RAW fence before re-reading P in A-fragment layout
    asm volatile("s_wait_dscnt 0x0" ::: "memory");
    v16h pf = load_a_frag(pw + nl * 40, half);
#pragma unroll
    for (int nt = 0; nt < 4; ++nt) {
      v16h vf = load_b_frag(Vs + (nt * 16 + nl) * 40, half);  // B[kk,dv]=V[kk,dv]
      o[nt] = WMMA16(pf, vf, o[nt]);
    }
  }

  // epilogue: O /= l ; layout [b, t, h*64 + dv] == concat-heads
#pragma unroll
  for (int v = 0; v < 8; ++v) {
    float inv = 1.f / lrow[v];
    _Float16* orow = O + (qrow0 + half * 8 + v) * 512 + col0;
#pragma unroll
    for (int nt = 0; nt < 4; ++nt)
      orow[nt * 16 + nl] = (_Float16)(o[nt][v] * inv);
  }
}

// ---------------------------------------------------------------------------
// Host side
// ---------------------------------------------------------------------------
static inline size_t align256(size_t x) { return (x + 255) & ~(size_t)255; }

extern "C" void kernel_launch(void* const* d_in, const int* in_sizes, int n_in,
                              void* d_out, int out_size, void* d_ws, size_t ws_size,
                              hipStream_t stream) {
  (void)in_sizes; (void)n_in; (void)out_size; (void)ws_size;
  constexpr int L = 6, D = 512, F = 2048, Bb = 8, T = 512, S = 1024;
  const int MT = Bb * T;   // 4096
  const int MS = Bb * S;   // 8192

  const float* tgt  = (const float*)d_in[0];
  const float* memf = (const float*)d_in[1];
  // d_in[2]/d_in[3]: source_mask (all ones) / target_mask (tril) -> analytic.
  const float* sa_w[4] = {(const float*)d_in[4],  (const float*)d_in[5],
                          (const float*)d_in[6],  (const float*)d_in[7]};
  const float* sa_b[4] = {(const float*)d_in[8],  (const float*)d_in[9],
                          (const float*)d_in[10], (const float*)d_in[11]};
  const float* ca_w[4] = {(const float*)d_in[12], (const float*)d_in[13],
                          (const float*)d_in[14], (const float*)d_in[15]};
  const float* ca_b[4] = {(const float*)d_in[16], (const float*)d_in[17],
                          (const float*)d_in[18], (const float*)d_in[19]};
  const float* ff_w1 = (const float*)d_in[20];
  const float* ff_b1 = (const float*)d_in[21];
  const float* ff_w2 = (const float*)d_in[22];
  const float* ff_b2 = (const float*)d_in[23];
  const float* ln_g  = (const float*)d_in[24];
  const float* ln_b  = (const float*)d_in[25];

  float* out32 = (float*)d_out;  // running `output` lives in d_out (f32)

  // workspace carve-up (f16 scratch, ~104 MB)
  char* cur = (char*)d_ws;
  auto take = [&](size_t elems) -> _Float16* {
    _Float16* p = (_Float16*)cur;
    cur += align256(elems * sizeof(_Float16));
    return p;
  };
  _Float16* w16_sa[4]; for (int j = 0; j < 4; ++j) w16_sa[j] = take((size_t)L * D * D);
  _Float16* w16_ca[4]; for (int j = 0; j < 4; ++j) w16_ca[j] = take((size_t)L * D * D);
  _Float16* w16_f1 = take((size_t)L * D * F);
  _Float16* w16_f2 = take((size_t)L * F * D);
  _Float16* mem16  = take((size_t)MS * D);
  _Float16* lnt16  = take((size_t)MT * D);
  _Float16* n16    = take((size_t)MT * D);
  _Float16* q16    = take((size_t)MT * D);
  _Float16* k16    = take((size_t)MS * D);
  _Float16* v16    = take((size_t)MS * D);
  _Float16* at16   = take((size_t)MT * D);
  _Float16* h16    = take((size_t)MT * F);

  auto cvt = [&](const float* src, _Float16* dst, size_t n) {
    cvt_f32_f16_k<<<2048, 256, 0, stream>>>(src, dst, n);
  };
  // prologue: f16 weights + memory; LN(tgt) once (SA input is identical
  // every layer -- faithful to the reference which never reassigns `tgt`)
  for (int j = 0; j < 4; ++j) {
    cvt(sa_w[j], w16_sa[j], (size_t)L * D * D);
    cvt(ca_w[j], w16_ca[j], (size_t)L * D * D);
  }
  cvt(ff_w1, w16_f1, (size_t)L * D * F);
  cvt(ff_w2, w16_f2, (size_t)L * F * D);
  cvt(memf, mem16, (size_t)MS * D);
  layernorm_k<true><<<MT / 8, 256, 0, stream>>>(tgt, ln_g, ln_b, lnt16, nullptr);

  const dim3 gT(MT / 128, D / 128);    // [4096 x 512]
  const dim3 gS(MS / 128, D / 128);    // [8192 x 512]
  const dim3 gF1(MT / 128, F / 128);   // [4096 x 2048]
  const dim3 gAttn(T / 64, Bb * 8);

  for (int i = 0; i < L; ++i) {
    const size_t wDD = (size_t)i * D * D, wDF = (size_t)i * D * F;
    const size_t wD = (size_t)i * D, wF = (size_t)i * F;

    // ---- self-attention (causal), input = LN(tgt), residual = tgt ----
    gemm_wmma_k<512, 512, false, false, true><<<gT, 256, 0, stream>>>(
        lnt16, w16_sa[0] + wDD, sa_b[0] + wD, nullptr, q16, nullptr);
    gemm_wmma_k<512, 512, false, false, true><<<gT, 256, 0, stream>>>(
        lnt16, w16_sa[1] + wDD, sa_b[1] + wD, nullptr, k16, nullptr);
    gemm_wmma_k<512, 512, false, false, true><<<gT, 256, 0, stream>>>(
        lnt16, w16_sa[2] + wDD, sa_b[2] + wD, nullptr, v16, nullptr);
    flash_attn_k<true, 512><<<gAttn, 128, 0, stream>>>(q16, k16, v16, at16);
    gemm_wmma_k<512, 512, false, true, false><<<gT, 256, 0, stream>>>(
        at16, w16_sa[3] + wDD, sa_b[3] + wD, tgt, nullptr, out32);
    layernorm_k<true><<<MT / 8, 256, 0, stream>>>(out32, ln_g, ln_b, n16, nullptr);

    // ---- cross-attention (no mask), K/V from raw memory ----
    gemm_wmma_k<512, 512, false, false, true><<<gT, 256, 0, stream>>>(
        n16, w16_ca[0] + wDD, ca_b[0] + wD, nullptr, q16, nullptr);
    gemm_wmma_k<512, 512, false, false, true><<<gS, 256, 0, stream>>>(
        mem16, w16_ca[1] + wDD, ca_b[1] + wD, nullptr, k16, nullptr);
    gemm_wmma_k<512, 512, false, false, true><<<gS, 256, 0, stream>>>(
        mem16, w16_ca[2] + wDD, ca_b[2] + wD, nullptr, v16, nullptr);
    flash_attn_k<false, 1024><<<gAttn, 128, 0, stream>>>(q16, k16, v16, at16);
    gemm_wmma_k<512, 512, false, true, false><<<gT, 256, 0, stream>>>(
        at16, w16_ca[3] + wDD, ca_b[3] + wD, out32, nullptr, out32);
    layernorm_k<true><<<MT / 8, 256, 0, stream>>>(out32, ln_g, ln_b, n16, nullptr);

    // ---- FFN ----
    gemm_wmma_k<2048, 512, true, false, true><<<gF1, 256, 0, stream>>>(
        n16, w16_f1 + wDF, ff_b1 + wF, nullptr, h16, nullptr);
    gemm_wmma_k<512, 2048, false, true, false><<<gT, 256, 0, stream>>>(
        h16, w16_f2 + wDF, ff_b2 + wD, out32, nullptr, out32);
  }

  // final shared LayerNorm, in place on d_out (one wave per row: all reads
  // of a row happen before any writes of that row)
  layernorm_k<false><<<MT / 8, 256, 0, stream>>>(out32, ln_g, ln_b, nullptr, out32);
}
